// AoAReader_1563368096451
// MI455X (gfx1250) — compile-verified, matrix operand order checked by reading, not run on hardware
//
#include <hip/hip_runtime.h>
#include <cstdint>
#include <cstddef>

// Problem constants
#define B_ 32
#define D_ 1024
#define Q_ 64
#define E_ 384
#define H_ 256
#define G_ 768   // 3*H
#define C_ 10

typedef __attribute__((ext_vector_type(16))) __bf16 v16bf;
typedef __attribute__((ext_vector_type(8)))  float  v8f;

// ---------------- helpers ----------------

static __device__ __forceinline__ unsigned short f2bf(float f) {
  unsigned int u = __float_as_uint(f);
  unsigned int r = (u + 0x7FFFu + ((u >> 16) & 1u)) >> 16;  // round-to-nearest-even
  return (unsigned short)r;
}

static __device__ __forceinline__ v8f wmma_bf16(v16bf a, v16bf b, v8f c) {
  return __builtin_amdgcn_wmma_f32_16x16x32_bf16(false, a, false, b, (short)0, c,
                                                 false, false);
}

static __device__ __forceinline__ v16bf ld16g(const unsigned short* p) {
  return *(const v16bf*)__builtin_assume_aligned((const void*)p, 32);
}

// WMMA fragment packing (ISA 7.12.2, 16-bit A 16x32 / mirrored B 32x16):
// element (mn, kk) -> lane = (mn&15) + 16*((kk>>3)&1), slot = (kk&7) | (((kk>>4)&1)<<3)
static __device__ __forceinline__ int packLane(int mn, int kk) {
  return (mn & 15) + 16 * ((kk >> 3) & 1);
}
static __device__ __forceinline__ int packSlot(int kk) {
  return (kk & 7) | (((kk >> 4) & 1) << 3);
}

// ---------------- kernel 1: pack weights (B layout, bf16) ----------------
// dst layout: [Kt][48][lane32][slot16], source w is (768, K) row-major fp32.
__global__ void pack_w_kernel(const float* __restrict__ w,
                              unsigned short* __restrict__ dst,
                              int Kt, int K) {
  int i = blockIdx.x * blockDim.x + threadIdx.x;
  int total = Kt * 48 * 512;
  if (i >= total) return;
  int slot = i & 15;
  int lane = (i >> 4) & 31;
  int nt   = (i >> 9) % 48;
  int kt   = (i >> 9) / 48;
  int nn = lane & 15;
  int lh = lane >> 4;
  int kk = (slot & 7) + 8 * lh + 16 * (slot >> 3);
  int n = nt * 16 + nn;
  int k = kt * 32 + kk;
  dst[i] = f2bf(w[(size_t)n * K + k]);
}

// ---------------- kernel 2: gather embeddings into A-packed bf16 ----------------
// rows m = b*T + t ; K = E = 384 (Kt = 12). dst: [Mt][12][lane32][slot16]
__global__ void pack_x_kernel(const int* __restrict__ tokens,
                              const int* __restrict__ lens,
                              const float* __restrict__ emb,
                              unsigned short* __restrict__ dst,
                              int T, int reversed) {
  int i = blockIdx.x * blockDim.x + threadIdx.x;
  int total = (B_ * T / 16) * 12 * 512;
  if (i >= total) return;
  int slot = i & 15;
  int lane = (i >> 4) & 31;
  int kt   = (i >> 9) % 12;
  int tm   = (i >> 9) / 12;
  int mm = lane & 15;
  int lh = lane >> 4;
  int kk = (slot & 7) + 8 * lh + 16 * (slot >> 3);
  int m = tm * 16 + mm;
  int e = kt * 32 + kk;
  int b = m / T;
  int t = m % T;
  if (reversed) {
    int L = lens[b];
    if (t < L) t = L - 1 - t;
  }
  int tok = tokens[b * T + t];
  dst[i] = f2bf(emb[(size_t)tok * E_ + e]);
}

// ---------------- kernel 3: gx = X * W_ih^T + b_ih  (WMMA, bf16->f32) ------------
// one wave computes a 16x64 strip; K = 384 (12 steps of 32). gx fp32 row-major N=768.
__global__ void gemm_gx_kernel(const unsigned short* __restrict__ Xp,
                               const unsigned short* __restrict__ Wb,
                               const float* __restrict__ bias,
                               float* __restrict__ gx, int Mt) {
  int wave = (blockIdx.x * blockDim.x + threadIdx.x) >> 5;
  int lane = threadIdx.x & 31;
  if (wave >= Mt * 12) return;  // wave-uniform
  int tm = wave / 12;
  int nq = wave % 12;
  v8f acc[4] = {v8f{}, v8f{}, v8f{}, v8f{}};
  for (int kt = 0; kt < 12; ++kt) {
    v16bf a = ld16g(Xp + (((size_t)tm * 12 + kt) * 32 + lane) * 16);
#pragma unroll
    for (int j = 0; j < 4; ++j) {
      int nt = nq * 4 + j;
      v16bf bb = ld16g(Wb + (((size_t)kt * 48 + nt) * 32 + lane) * 16);
      acc[j] = wmma_bf16(a, bb, acc[j]);
    }
  }
  int nn = lane & 15;
  int rh = (lane >> 4) * 8;
#pragma unroll
  for (int j = 0; j < 4; ++j) {
    int n = (nq * 4 + j) * 16 + nn;
    float bv = bias[n];
#pragma unroll
    for (int r = 0; r < 8; ++r) {
      int m = tm * 16 + rh + r;
      gx[(size_t)m * G_ + n] = acc[j][r] + bv;
    }
  }
}

// ---------------- kernel 4: GRU recurrence (4 independent blocks) ----------------
// block 0: doc fwd, 1: doc bwd, 2: query fwd, 3: query bwd. 1024 threads = 32 waves.
// Per step: gh = h * W_hh^T via WMMA (96 tiles, 3/wave), then elementwise GRU update.
__global__ __launch_bounds__(1024, 1) void gru_kernel(
    const float* __restrict__ gx_df, const float* __restrict__ gx_db,
    const float* __restrict__ gx_qf, const float* __restrict__ gx_qb,
    const unsigned short* __restrict__ whh_f, const unsigned short* __restrict__ whh_b,
    const float* __restrict__ bhh_f, const float* __restrict__ bhh_b,
    const int* __restrict__ dlen, const int* __restrict__ qlen,
    unsigned short* __restrict__ dosP, unsigned short* __restrict__ qosP) {
  __shared__ __align__(32) float gh[32 * G_];            // 96 KB
  __shared__ __align__(32) float hcur[32 * H_];          // 32 KB
  __shared__ __align__(32) unsigned short hA[2 * 8 * 512];  // 16 KB, A-packed h

  int cfg = blockIdx.x;
  int isDoc = (cfg < 2) ? 1 : 0;
  int rev = cfg & 1;
  int T = isDoc ? D_ : Q_;
  const float* gx = (cfg == 0) ? gx_df : (cfg == 1) ? gx_db : (cfg == 2) ? gx_qf : gx_qb;
  const unsigned short* whh = rev ? whh_b : whh_f;
  const float* bhh = rev ? bhh_b : bhh_f;
  const int* lens = isDoc ? dlen : qlen;
  int kofs = rev ? H_ : 0;

  int tid = threadIdx.x;
  int lane = tid & 31;
  int wave = tid >> 5;

  for (int i = tid; i < 32 * H_; i += 1024) hcur[i] = 0.0f;
  for (int i = tid; i < 2 * 8 * 512; i += 1024) hA[i] = 0;
  __syncthreads();

  for (int t = 0; t < T; ++t) {
    // --- gh = h * W_hh^T : 2x48 C tiles, wave handles 3 ---
#pragma unroll
    for (int q = 0; q < 3; ++q) {
      int ct = wave * 3 + q;
      int tm = ct / 48;
      int nt = ct % 48;
      v8f acc = v8f{};
#pragma unroll
      for (int kt = 0; kt < 8; ++kt) {
        v16bf a = *(const v16bf*)&hA[((tm * 8 + kt) * 32 + lane) * 16];
        v16bf bb = ld16g(whh + (((size_t)kt * 48 + nt) * 32 + lane) * 16);
        acc = wmma_bf16(a, bb, acc);
      }
      int nn = lane & 15;
      int rh = (lane >> 4) * 8;
#pragma unroll
      for (int r = 0; r < 8; ++r)
        gh[(tm * 16 + rh + r) * G_ + nt * 16 + nn] = acc[r];
    }
    __syncthreads();

    // --- elementwise GRU update: 32*256 = 8192 states ---
    for (int e = tid; e < 32 * H_; e += 1024) {
      int b = e >> 8;
      int j = e & 255;
      const float* gxr = gx + (size_t)(b * T + t) * G_;
      float rg = gxr[j] + gh[b * G_ + j] + bhh[j];
      rg = 1.0f / (1.0f + __expf(-rg));
      float zg = gxr[H_ + j] + gh[b * G_ + H_ + j] + bhh[H_ + j];
      zg = 1.0f / (1.0f + __expf(-zg));
      float ng = tanhf(gxr[2 * H_ + j] + rg * (gh[b * G_ + 2 * H_ + j] + bhh[2 * H_ + j]));
      float hn = (1.0f - zg) * ng + zg * hcur[e];
      hcur[e] = hn;
      unsigned short hb = f2bf(hn);

      // repack h into WMMA A-layout for the next step
      int tm2 = b >> 4, tk = j >> 5, kk = j & 31;
      hA[((tm2 * 8 + tk) * 32 + packLane(b, kk)) * 16 + packSlot(kk)] = hb;

      // write output to packed dos (A layout) or qos (B layout)
      int p = t;
      if (rev) {
        int L = lens[b];
        p = (t < L) ? (L - 1 - t) : t;
      }
      int kc = kofs + j;
      int tko = kc >> 5, kko = kc & 31;
      if (isDoc) {
        int tmo = p >> 4;
        dosP[(size_t)b * 524288 +
             (((size_t)tmo * 16 + tko) * 32 + packLane(p, kko)) * 16 + packSlot(kko)] = hb;
      } else {
        int nto = p >> 4;
        qosP[(size_t)b * 32768 +
             (((size_t)tko * 4 + nto) * 32 + packLane(p, kko)) * 16 + packSlot(kko)] = hb;
      }
    }
    __syncthreads();
  }
}

// ---------------- kernel 5: M[b,d,q] = dos . qos  (WMMA) ----------------
// wave per (b, d-tile): 16x64 output, K = 512 (16 steps).
__global__ void attn_gemm_kernel(const unsigned short* __restrict__ dosP,
                                 const unsigned short* __restrict__ qosP,
                                 float* __restrict__ Mout) {
  int wave = (blockIdx.x * blockDim.x + threadIdx.x) >> 5;
  int lane = threadIdx.x & 31;
  if (wave >= B_ * 64) return;
  int b = wave >> 6;
  int tm = wave & 63;
  v8f acc[4] = {v8f{}, v8f{}, v8f{}, v8f{}};
  for (int kt = 0; kt < 16; ++kt) {
    v16bf a = ld16g(dosP + (size_t)b * 524288 + (((size_t)tm * 16 + kt) * 32 + lane) * 16);
#pragma unroll
    for (int j = 0; j < 4; ++j) {
      v16bf bb = ld16g(qosP + (size_t)b * 32768 + (((size_t)kt * 4 + j) * 32 + lane) * 16);
      acc[j] = wmma_bf16(a, bb, acc[j]);
    }
  }
  int nn = lane & 15;
  int rh = (lane >> 4) * 8;
#pragma unroll
  for (int j = 0; j < 4; ++j) {
#pragma unroll
    for (int r = 0; r < 8; ++r) {
      int d = tm * 16 + rh + r;
      int q = j * 16 + nn;
      Mout[((size_t)b * D_ + d) * Q_ + q] = acc[j][r];
    }
  }
}

// ---------------- kernel 6: alpha = softmax over d (axis=1), masked ------------
__global__ void alpha_kernel(const float* __restrict__ Mm,
                             const float* __restrict__ dmask,
                             const float* __restrict__ qmask,
                             float* __restrict__ alpha) {
  int b = blockIdx.x >> 6;
  int q = blockIdx.x & 63;
  int tid = threadIdx.x;
  __shared__ float red[256];

  float mx = -3.402823466e38f;
  for (int d = tid; d < D_; d += 256)
    mx = fmaxf(mx, Mm[((size_t)b * D_ + d) * Q_ + q]);
  red[tid] = mx;
  __syncthreads();
  for (int s2 = 128; s2 > 0; s2 >>= 1) {
    if (tid < s2) red[tid] = fmaxf(red[tid], red[tid + s2]);
    __syncthreads();
  }
  mx = red[0];
  __syncthreads();

  float qm = qmask[b * Q_ + q];
  float sum = 0.0f;
  for (int d = tid; d < D_; d += 256) {
    size_t idx = ((size_t)b * D_ + d) * Q_ + q;
    float e = __expf(Mm[idx] - mx) * dmask[b * D_ + d] * qm;
    alpha[idx] = e;
    sum += e;
  }
  red[tid] = sum;
  __syncthreads();
  for (int s2 = 128; s2 > 0; s2 >>= 1) {
    if (tid < s2) red[tid] += red[tid + s2];
    __syncthreads();
  }
  float inv = 1.0f / (red[0] + 1e-12f);
  for (int d = tid; d < D_; d += 256) {
    size_t idx = ((size_t)b * D_ + d) * Q_ + q;
    alpha[idx] *= inv;
  }
}

// ---------------- kernel 7: per-row (b,d) stats for beta ----------------
__global__ void row_stats_kernel(const float* __restrict__ Mm,
                                 const float* __restrict__ dmask,
                                 const float* __restrict__ qmask,
                                 float* __restrict__ rmax,
                                 float* __restrict__ rinv) {
  int i = blockIdx.x * blockDim.x + threadIdx.x;
  if (i >= B_ * D_) return;
  int b = i >> 10;
  const float* row = Mm + (size_t)i * Q_;
  float mx = -3.402823466e38f;
  for (int q = 0; q < Q_; ++q) mx = fmaxf(mx, row[q]);
  float dm = dmask[i];
  float sum = 0.0f;
  for (int q = 0; q < Q_; ++q) sum += __expf(row[q] - mx) * qmask[b * Q_ + q];
  sum *= dm;
  rmax[i] = mx;
  rinv[i] = 1.0f / (sum + 1e-12f);
}

// ---------------- kernel 8: avg_beta[b,q] = (sum_d beta)/len ----------------
__global__ void avgbeta_kernel(const float* __restrict__ Mm,
                               const float* __restrict__ dmask,
                               const float* __restrict__ qmask,
                               const float* __restrict__ rmax,
                               const float* __restrict__ rinv,
                               const int* __restrict__ dlen,
                               float* __restrict__ avgb) {
  int b = blockIdx.x >> 6;
  int q = blockIdx.x & 63;
  int tid = threadIdx.x;
  __shared__ float red[256];
  float part = 0.0f;
  for (int d = tid; d < D_; d += 256) {
    int i = b * D_ + d;
    part += __expf(Mm[(size_t)i * Q_ + q] - rmax[i]) * dmask[i] * rinv[i];
  }
  red[tid] = part;
  __syncthreads();
  for (int s2 = 128; s2 > 0; s2 >>= 1) {
    if (tid < s2) red[tid] += red[tid + s2];
    __syncthreads();
  }
  if (tid == 0)
    avgb[b * Q_ + q] = red[0] * qmask[b * Q_ + q] / (float)dlen[b];
}

// ---------------- kernel 9: s[b,d] = sum_q alpha * avg_beta ----------------
__global__ void s_kernel(const float* __restrict__ alpha,
                         const float* __restrict__ avgb,
                         float* __restrict__ s) {
  int i = blockIdx.x * blockDim.x + threadIdx.x;
  if (i >= B_ * D_) return;
  int b = i >> 10;
  const float* arow = alpha + (size_t)i * Q_;
  const float* ab = avgb + b * Q_;
  float acc = 0.0f;
  for (int q = 0; q < Q_; ++q) acc += arow[q] * ab[q];
  s[i] = acc;
}

// ---------------- kernel 10: candidate scoring + argmax + probs ----------------
__global__ void final_kernel(const float* __restrict__ s,
                             const int* __restrict__ docs,
                             const int* __restrict__ cands,
                             const int* __restrict__ ans,
                             float* __restrict__ out) {
  int b = blockIdx.x;
  int tid = threadIdx.x;
  __shared__ float red[256];
  __shared__ float res[11];

  int cd[10];
#pragma unroll
  for (int c = 0; c < 10; ++c) cd[c] = cands[b * C_ + c];
  int av = ans[b];

  float lc[11];
#pragma unroll
  for (int c = 0; c < 11; ++c) lc[c] = 0.0f;

  for (int d = tid; d < D_; d += 256) {
    float sv = s[b * D_ + d];
    int tok = docs[b * D_ + d];
#pragma unroll
    for (int c = 0; c < 10; ++c)
      if (tok == cd[c]) lc[c] += sv;
    if (tok == av) lc[10] += sv;
  }

#pragma unroll
  for (int c = 0; c < 11; ++c) {
    red[tid] = lc[c];
    __syncthreads();
    for (int s2 = 128; s2 > 0; s2 >>= 1) {
      if (tid < s2) red[tid] += red[tid + s2];
      __syncthreads();
    }
    if (tid == 0) res[c] = red[0];
    __syncthreads();
  }

  if (tid == 0) {
    int best = 0;
    float bv = res[0];
#pragma unroll
    for (int c = 1; c < 10; ++c)
      if (res[c] > bv) { bv = res[c]; best = c; }
    out[b] = (float)cd[best];        // pred_answers
    out[B_ + b] = (float)best;       // pred_locs
    out[2 * B_ + b] = res[10];       // probs
  }
}

// ---------------- host launcher ----------------
extern "C" void kernel_launch(void* const* d_in, const int* in_sizes, int n_in,
                              void* d_out, int out_size, void* d_ws, size_t ws_size,
                              hipStream_t stream) {
  (void)in_sizes; (void)n_in; (void)out_size; (void)ws_size;

  const int*   docs_input  = (const int*)d_in[0];
  const int*   docs_len    = (const int*)d_in[1];
  const float* doc_mask    = (const float*)d_in[2];
  const int*   querys_input= (const int*)d_in[3];
  const int*   querys_len  = (const int*)d_in[4];
  const float* query_mask  = (const float*)d_in[5];
  const int*   candidates  = (const int*)d_in[6];
  const int*   answers     = (const int*)d_in[7];
  const float* embedding   = (const float*)d_in[8];
  const float* w_ih_f      = (const float*)d_in[9];
  const float* w_hh_f      = (const float*)d_in[10];
  const float* b_ih_f      = (const float*)d_in[11];
  const float* b_hh_f      = (const float*)d_in[12];
  const float* w_ih_b      = (const float*)d_in[13];
  const float* w_hh_b      = (const float*)d_in[14];
  const float* b_ih_b      = (const float*)d_in[15];
  const float* b_hh_b      = (const float*)d_in[16];
  float* out = (float*)d_out;

  char* ws = (char*)d_ws;
  // byte offsets (all 256-aligned)
  const size_t OFF_WIHF = 0;                 // 12*48*512*2 = 589824
  const size_t OFF_WIHB = 589824;
  const size_t OFF_WHHF = 1179648;           // 8*48*512*2 = 393216
  const size_t OFF_WHHB = 1572864;
  const size_t OFF_XDF  = 1966080;           // 2048*12*512*2 = 25165824
  const size_t OFF_XDB  = 27131904;
  const size_t OFF_XQF  = 52297728;          // 128*12*512*2 = 1572864
  const size_t OFF_XQB  = 53870592;
  const size_t OFF_GXDF = 55443456;          // 32768*768*4 = 100663296
  const size_t OFF_GXDB = 156106752;
  const size_t OFF_GXQF = 256770048;         // 2048*768*4 = 6291456
  const size_t OFF_GXQB = 263061504;
  const size_t OFF_DOSP = 269352960;         // 32*524288*2 = 33554432
  const size_t OFF_QOSP = 302907392;         // 32*32768*2 = 2097152
  const size_t OFF_M    = 305004544;         // 8388608
  const size_t OFF_AL   = 313393152;         // 8388608
  const size_t OFF_AVGB = 321781760;         // 8192
  const size_t OFF_S    = 321789952;         // 131072
  const size_t OFF_RMAX = 321921024;         // 131072
  const size_t OFF_RINV = 322052096;         // 131072

  unsigned short* WihF = (unsigned short*)(ws + OFF_WIHF);
  unsigned short* WihB = (unsigned short*)(ws + OFF_WIHB);
  unsigned short* WhhF = (unsigned short*)(ws + OFF_WHHF);
  unsigned short* WhhB = (unsigned short*)(ws + OFF_WHHB);
  unsigned short* XDF  = (unsigned short*)(ws + OFF_XDF);
  unsigned short* XDB  = (unsigned short*)(ws + OFF_XDB);
  unsigned short* XQF  = (unsigned short*)(ws + OFF_XQF);
  unsigned short* XQB  = (unsigned short*)(ws + OFF_XQB);
  float* GXDF = (float*)(ws + OFF_GXDF);
  float* GXDB = (float*)(ws + OFF_GXDB);
  float* GXQF = (float*)(ws + OFF_GXQF);
  float* GXQB = (float*)(ws + OFF_GXQB);
  unsigned short* DOSP = (unsigned short*)(ws + OFF_DOSP);
  unsigned short* QOSP = (unsigned short*)(ws + OFF_QOSP);
  float* MM   = (float*)(ws + OFF_M);
  float* AL   = (float*)(ws + OFF_AL);
  float* AVGB = (float*)(ws + OFF_AVGB);
  float* SS   = (float*)(ws + OFF_S);
  float* RMAX = (float*)(ws + OFF_RMAX);
  float* RINV = (float*)(ws + OFF_RINV);

  // 1) pack weights to bf16 WMMA-B layout
  pack_w_kernel<<<1152, 256, 0, stream>>>(w_ih_f, WihF, 12, E_);
  pack_w_kernel<<<1152, 256, 0, stream>>>(w_ih_b, WihB, 12, E_);
  pack_w_kernel<<<768,  256, 0, stream>>>(w_hh_f, WhhF, 8, H_);
  pack_w_kernel<<<768,  256, 0, stream>>>(w_hh_b, WhhB, 8, H_);

  // 2) gather embeddings into bf16 WMMA-A layout (bwd folds rev_within_len)
  pack_x_kernel<<<49152, 256, 0, stream>>>(docs_input, docs_len, embedding, XDF, D_, 0);
  pack_x_kernel<<<49152, 256, 0, stream>>>(docs_input, docs_len, embedding, XDB, D_, 1);
  pack_x_kernel<<<3072,  256, 0, stream>>>(querys_input, querys_len, embedding, XQF, Q_, 0);
  pack_x_kernel<<<3072,  256, 0, stream>>>(querys_input, querys_len, embedding, XQB, Q_, 1);

  // 3) input projections gx = X*W_ih^T + b_ih  (WMMA)
  gemm_gx_kernel<<<3072, 256, 0, stream>>>(XDF, WihF, b_ih_f, GXDF, 2048);
  gemm_gx_kernel<<<3072, 256, 0, stream>>>(XDB, WihB, b_ih_b, GXDB, 2048);
  gemm_gx_kernel<<<192,  256, 0, stream>>>(XQF, WihF, b_ih_f, GXQF, 128);
  gemm_gx_kernel<<<192,  256, 0, stream>>>(XQB, WihB, b_ih_b, GXQB, 128);

  // 4) four concurrent GRU recurrences (doc f/b, query f/b), WMMA per step
  gru_kernel<<<4, 1024, 0, stream>>>(GXDF, GXDB, GXQF, GXQB, WhhF, WhhB,
                                     b_hh_f, b_hh_b, docs_len, querys_len,
                                     DOSP, QOSP);

  // 5) attention scores M = dos . qos^T  (WMMA)
  attn_gemm_kernel<<<256, 256, 0, stream>>>(DOSP, QOSP, MM);

  // 6-8) masked softmaxes
  alpha_kernel<<<B_ * Q_, 256, 0, stream>>>(MM, doc_mask, query_mask, AL);
  row_stats_kernel<<<128, 256, 0, stream>>>(MM, doc_mask, query_mask, RMAX, RINV);
  avgbeta_kernel<<<B_ * Q_, 256, 0, stream>>>(MM, doc_mask, query_mask, RMAX, RINV,
                                              docs_len, AVGB);

  // 9) s = alpha . avg_beta
  s_kernel<<<128, 256, 0, stream>>>(AL, AVGB, SS);

  // 10) candidate scoring, argmax, probs
  final_kernel<<<B_, 256, 0, stream>>>(SS, docs_input, candidates, answers, out);
}
